// Prediction_3032246911362
// MI455X (gfx1250) — matrix operand order repaired and checked by reading, not compiled, not run on hardware
//
#include <hip/hip_runtime.h>

typedef __attribute__((ext_vector_type(16))) _Float16 v16h;
typedef __attribute__((ext_vector_type(8)))  float    v8f;

#define IMG_H 256
#define IMG_W 256
#define TILE  16
#define SPAN  21          // search_size
#define PADR  10          // search_size/2
#define NS    441         // SPAN*SPAN
#define NSP   448         // padded to 14*32 for WMMA K-chunks
#define WINW  36          // TILE + 2*PADR
#define WINR  37          // +1 zero row for padded shifts (i==21)
#define NTHREADS 512

// ---- CDNA5 async global->LDS copy (gfx1250 VGLOBAL async op, ASYNCcnt) ----
__device__ __forceinline__ void asyncLoadF32(unsigned ldsAddr, const float* g) {
    asm volatile("global_load_async_to_lds_b32 %0, %1, off"
                 :: "v"(ldsAddr), "v"(g) : "memory");
}
__device__ __forceinline__ void waitAsync() {
    asm volatile("s_wait_asynccnt 0" ::: "memory");
}
__device__ __forceinline__ unsigned ldsOff(const void* p) {
    return (unsigned)(unsigned long long)p;   // low 32 bits of generic ptr = LDS offset
}

// ---- wave32 reductions ----
__device__ __forceinline__ float waveMin(float v) {
#pragma unroll
    for (int m = 16; m >= 1; m >>= 1) v = fminf(v, __shfl_xor(v, m, 32));
    return v;
}
__device__ __forceinline__ float waveSum(float v) {
#pragma unroll
    for (int m = 16; m >= 1; m >>= 1) v += __shfl_xor(v, m, 32);
    return v;
}

__global__ __launch_bounds__(NTHREADS)
void interpred_kernel(const float* __restrict__ im1,
                      const float* __restrict__ im2,
                      float* __restrict__ out) {
    __shared__ float im2win[WINR][WINW];                            // 5328 B
    __shared__ float im1T[TILE][TILE];                              // 1024 B
    __shared__ _Float16 wBlk[16][NSP] __attribute__((aligned(32))); // 14336 B
    __shared__ unsigned offTab[NSP];                                // 1792 B

    const int tid  = threadIdx.x;
    const int lane = tid & 31;
    const int wave = tid >> 5;           // one wave per 4x4 block (16 blocks/tile)

    const int batch = blockIdx.x >> 8;   // 256 tiles per batch image
    const int rem   = blockIdx.x & 255;
    const int gy0   = (rem >> 4) * TILE;
    const int gx0   = (rem & 15) * TILE;

    const float* im1b = im1 + batch * (IMG_H * IMG_W);
    const float* im2b = im2 + batch * (IMG_H * IMG_W);

    // ---- zero the window (covers zero-padding and the extra pad row) ----
    for (int t = tid; t < WINR * WINW; t += NTHREADS)
        ((float*)im2win)[t] = 0.0f;
    // ---- shift -> window byte-offset table (replaces per-element div/mod 21) ----
    if (tid < NSP) {
        int i = tid / 21, j = tid - i * 21;
        offTab[tid] = (unsigned)((i * WINW + j) * 4);
    }
    __syncthreads();

    // ---- async-stage im2 halo window + im1 tile into LDS ----
    for (int t = tid; t < WINW * WINW; t += NTHREADS) {
        int r = t / WINW, c = t - r * WINW;
        int gy = gy0 - PADR + r, gx = gx0 - PADR + c;
        if ((unsigned)gy < IMG_H && (unsigned)gx < IMG_W)
            asyncLoadF32(ldsOff(&im2win[r][c]), im2b + gy * IMG_W + gx);
    }
    for (int t = tid; t < TILE * TILE; t += NTHREADS) {
        int r = t >> 4, c = t & 15;
        asyncLoadF32(ldsOff(&im1T[r][c]), im1b + (gy0 + r) * IMG_W + gx0 + c);
    }
    waitAsync();
    __syncthreads();

    // ---- per-wave block origin inside the tile ----
    const int ty = (wave >> 2) << 2;
    const int tx = (wave & 3) << 2;
    const char* winBytes = (const char*)&im2win[0][0];
    const unsigned blkBase = (unsigned)((ty * WINW + tx) * 4);

    float im1v[16];
#pragma unroll
    for (int p = 0; p < 16; ++p)
        im1v[p] = im1T[ty + (p >> 2)][tx + (p & 3)];

    // ---- SAD volume: lane l owns shifts s = 32*cc + l ----
    float vols[14];
    float vmin = 3.0e38f;
#pragma unroll
    for (int cc = 0; cc < 14; ++cc) {
        int s = cc * 32 + lane;
        const float* bp = (const float*)(winBytes + (blkBase + offTab[s]));
        float acc = 0.0f;
#pragma unroll
        for (int p = 0; p < 16; ++p) {
            float d = im1v[p] - bp[(p >> 2) * WINW + (p & 3)];
            acc += fabsf(d);
        }
        bool valid = (s < NS);
        vols[cc] = valid ? acc * (1.0f / 16.0f) : 3.0e38f;
        if (valid) vmin = fminf(vmin, vols[cc]);
    }
    vmin = waveMin(vmin);

    // ---- softmax(-100*vol): exp(100*(vmin - vol)) / sum ----
    float ev[14];
    float esum = 0.0f;
#pragma unroll
    for (int cc = 0; cc < 14; ++cc) {
        float e = (vols[cc] < 1.0e38f)
                ? exp2f(144.26950408889634f * (vmin - vols[cc])) : 0.0f;
        ev[cc] = e;
        esum  += e;
    }
    esum = waveSum(esum);
    const float inv = 1.0f / esum;
#pragma unroll
    for (int cc = 0; cc < 14; ++cc) {
        int s = cc * 32 + lane;
        wBlk[wave][s] = (_Float16)(ev[cc] * inv);   // zeros for padded s >= 441
    }
    __syncthreads();

    // ---- weighted sum over shifts via v_wmma_f32_16x16x32_f16 ----
    // A (16x32): row M = block pixel, col K = shift within chunk.
//   lane<16 : M=lane,    halves 0..15 -> K=0..15
//   lane>=16: M=lane-16, halves 0..15 -> K=16..31
    // B (32x16): B[k][n] = w[k] for all n (broadcast across N columns).
    //   lane<16 : half h -> K=h ; lane>=16: half h -> K=16+h
    const int p    = lane & 15;
    const int rowb = ty + (p >> 2), colb = tx + (p & 3);
    const int aK   = (lane & 16) ? 16 : 0;
    const int bK   = (lane & 16) ? 16 : 0;
    const unsigned pixBase = (unsigned)((rowb * WINW + colb) * 4);

    v8f cacc = {};
#pragma unroll
    for (int cc = 0; cc < 14; ++cc) {
        const int k0 = cc * 32;
        v16h a;
#pragma unroll
        for (int h = 0; h < 16; ++h) {
            int K = h + aK;
            unsigned off = offTab[k0 + K];       // one contiguous 16-entry run
            float v = *(const float*)(winBytes + (pixBase + off));
            a[h] = (_Float16)v;
        }
        v16h bfrag = *(const v16h*)&wBlk[wave][k0 + bK];
        cacc = __builtin_amdgcn_wmma_f32_16x16x32_f16(
            /*neg_a=*/false, a, /*neg_b=*/false, bfrag,
            /*c_mod=*/(short)0, cacc, /*reuse_a=*/false, /*reuse_b=*/false);
    }

    // D layout: VGPR e -> M=e (lanes 0-15) / M=8+e (lanes 16-31); all N columns
    // are identical (B broadcast), so lanes 0 and 16 write the 16 block pixels.
    if ((lane & 15) == 0) {
        const int mbase = (lane >> 4) * 8;
#pragma unroll
        for (int e = 0; e < 8; ++e) {
            int m = mbase + e;
            int y = gy0 + ty + (m >> 2);
            int x = gx0 + tx + (m & 3);
            out[(batch * IMG_H + y) * IMG_W + x] = cacc[e];
        }
    }
}

extern "C" void kernel_launch(void* const* d_in, const int* in_sizes, int n_in,
                              void* d_out, int out_size, void* d_ws, size_t ws_size,
                              hipStream_t stream) {
    const float* im1 = (const float*)d_in[0];
    const float* im2 = (const float*)d_in[1];
    float* out = (float*)d_out;
    // 2 batches * (256/16)^2 tiles = 512 workgroups, 512 threads (16 wave32s) each
    dim3 grid(512), block(NTHREADS);
    hipLaunchKernelGGL(interpred_kernel, grid, block, 0, stream, im1, im2, out);
}
